// Sparse_conv2d_87806311399783
// MI455X (gfx1250) — matrix operand order repaired and checked by reading, not compiled, not run on hardware
//
#include <hip/hip_runtime.h>

// ---------------------------------------------------------------------------
// Sparse conv2d as gather + GEMM on gfx1250 (MI455X).
//   out[256, 65536] = W[256, 2304] @ G[2304, 65536]
// G gathered from padded feature map via (h_idx, w_idx).
// bf16 WMMA (v_wmma_f32_16x16x32_bf16), f32 accumulation,
// double-buffered LDS B tile with software-pipelined gather.
// ---------------------------------------------------------------------------

typedef __bf16 v16bf __attribute__((ext_vector_type(16)));
typedef float  v8f   __attribute__((ext_vector_type(8)));

#define H_    512
#define W_    512
#define CIN   256
#define COUT  256
#define NPTS  65536
#define KTOT  2304          // CIN * 9
#define HW    (H_ * W_)

#define NTILE   64          // output columns per block
#define KSTEP   32          // bf16 WMMA K depth
#define BSTRIDE 40          // padded bf16 row stride for B tile (80 B, 16-B aligned)

__device__ __forceinline__ unsigned short f32_to_bf16_rne(float f) {
    unsigned int u = __float_as_uint(f);
    u += 0x7FFFu + ((u >> 16) & 1u);          // round-to-nearest-even
    return (unsigned short)(u >> 16);
}

// --- Kernel 1: weights f32 -> bf16 into workspace (row-major [co][k]) ------
__global__ void wconv_bf16_kernel(const float* __restrict__ w,
                                  unsigned short* __restrict__ wb) {
    int i = blockIdx.x * blockDim.x + threadIdx.x;
    if (i < COUT * KTOT) wb[i] = f32_to_bf16_rne(w[i]);
}

// --- Kernel 2: gather + bf16 WMMA GEMM -------------------------------------
// Block: 512 threads (16 waves). Each block: all 256 out-channels x 64 cols.
// Wave w computes out-channel rows [w*16, w*16+16) over 4 column groups of 16.
__launch_bounds__(512)
__global__ void spconv_wmma_kernel(const float* __restrict__ feat,
                                   const unsigned short* __restrict__ wbf,
                                   const int* __restrict__ hidx,
                                   const int* __restrict__ widx,
                                   float* __restrict__ out) {
    __shared__ int            sOff[9 * NTILE];             // gather offsets (-1 = pad)
    __shared__ unsigned short sB[2][NTILE * BSTRIDE];      // double-buffered B tile

    const int t    = threadIdx.x;
    const int lane = t & 31;
    const int wave = t >> 5;            // 0..15
    const int hi   = lane >> 4;         // 0/1 (lane half)
    const int ln   = lane & 15;         // A row (M) / B,C column (N) within subtile

    const int colBase = blockIdx.x * NTILE;

    // ---- precompute per-column gather offsets for the 9 taps --------------
    for (int e = t; e < 9 * NTILE; e += 512) {
        int p   = e >> 6;               // tap 0..8
        int col = e & 63;
        int j   = colBase + col;
        int fh  = hidx[p * NPTS + j] - 1;   // padded -> feature coords
        int fw  = widx[p * NPTS + j] - 1;
        bool ok = ((unsigned)fh < (unsigned)H_) && ((unsigned)fw < (unsigned)W_);
        sOff[e] = ok ? (fh * W_ + fw) : -1;
    }
    __syncthreads();                    // sOff ready for the pipelined gathers

    // staging role: 8 threads per column, each stages 4 consecutive k values
    const int sCol = t >> 3;            // 0..63
    const int sKr  = (t & 7) * 4;       // 0,4,...,28

    // issue the 4 random gather loads for one K step (values land in regs)
    auto gather4 = [&](int kb, float vv[4]) {
        #pragma unroll
        for (int i = 0; i < 4; ++i) {
            int k   = kb + sKr + i;
            int ci  = k / 9;
            int p   = k - 9 * ci;
            int off = sOff[p * NTILE + sCol];
            vv[i] = (off >= 0) ? feat[(size_t)ci * HW + off] : 0.0f;
        }
    };
    // convert + pack + ds_store_b64 into buffer `buf`
    auto store4 = [&](int buf, const float vv[4]) {
        uint2 u2;
        u2.x = (unsigned)f32_to_bf16_rne(vv[0]) | ((unsigned)f32_to_bf16_rne(vv[1]) << 16);
        u2.y = (unsigned)f32_to_bf16_rne(vv[2]) | ((unsigned)f32_to_bf16_rne(vv[3]) << 16);
        *(uint2*)(&sB[buf][sCol * BSTRIDE + sKr]) = u2;    // 8-B aligned
    };

    // A fragment: 16x32 bf16, rows = wave*16 + ln.
    // Per-lane K layout (ISA 7.12.2): e=0..7 -> K=8hi+e ; e=8..15 -> K=16+8hi+(e-8)
    union Afrag { uint4 q[2]; v16bf v; };
    auto loadA = [&](int kb, Afrag& A) {
        const unsigned short* wrow =
            wbf + (size_t)(wave * 16 + ln) * KTOT + kb + 8 * hi;
        A.q[0] = *(const uint4*)(wrow);        // K = 8hi .. 8hi+7     (b128)
        A.q[1] = *(const uint4*)(wrow + 16);   // K = 16+8hi .. +7     (b128)
    };

    v8f acc[4] = {v8f{}, v8f{}, v8f{}, v8f{}};

    // ---- pipeline prologue: stage step 0 ----------------------------------
    float cur[4];
    gather4(0, cur);
    store4(0, cur);
    Afrag Acur;
    loadA(0, Acur);

    // ---- main K loop: 1 barrier per step, gather overlapped with WMMA -----
    int buf = 0;
    for (int kb = 0; kb < KTOT; kb += KSTEP, buf ^= 1) {
        const bool more = (kb + KSTEP) < KTOT;

        float nxt[4];
        Afrag Anxt;
        if (more) {
            gather4(kb + KSTEP, nxt);      // random loads fly over the WMMAs
            loadA(kb + KSTEP, Anxt);
        }

        __syncthreads();                   // buffer `buf` fully staged

        #pragma unroll
        for (int g = 0; g < 4; ++g) {
            // B 32x16 bf16: lane holds column N=ln, e=0..7 -> K=8hi+e ; e=8..15 -> K=16+8hi+(e-8)
            union { uint4 q[2]; v16bf v; } B;
            const unsigned short* brow = &sB[buf][(g * 16 + ln) * BSTRIDE + 8 * hi];
            B.q[0] = *(const uint4*)(brow);        // ds_load_b128
            B.q[1] = *(const uint4*)(brow + 16);   // ds_load_b128
            acc[g] = __builtin_amdgcn_wmma_f32_16x16x32_bf16(
                /*neg_a=*/false, Acur.v, /*neg_b=*/false, B.v,
                /*c_mod=*/(short)0, acc[g],
                /*reuse_a=*/false, /*reuse_b=*/false);
        }

        if (more) {
            store4(buf ^ 1, nxt);          // fill the other buffer for next step
            Acur = Anxt;
        }
    }

    // ---- write C/D: VGPR v -> M = v + 8*hi, N = ln ------------------------
    const int coBase = wave * 16;
    #pragma unroll
    for (int g = 0; g < 4; ++g) {
        int j = colBase + g * 16 + ln;
        #pragma unroll
        for (int v = 0; v < 8; ++v) {
            int m = v + 8 * hi;
            out[(size_t)(coBase + m) * NPTS + j] = acc[g][v];
        }
    }
}

extern "C" void kernel_launch(void* const* d_in, const int* in_sizes, int n_in,
                              void* d_out, int out_size, void* d_ws, size_t ws_size,
                              hipStream_t stream) {
    const float* feat = (const float*)d_in[0];   // (1, 256, 512, 512) f32
    const float* wgt  = (const float*)d_in[1];   // (256, 256, 3, 3)   f32
    const int*   hidx = (const int*)d_in[2];     // (9, 65536) i32
    const int*   widx = (const int*)d_in[3];     // (9, 65536) i32
    float*       outp = (float*)d_out;           // (256, 65536) f32

    unsigned short* wbf = (unsigned short*)d_ws; // 256*2304 bf16 = 1.125 MB

    // 1) weights -> bf16
    {
        int n = COUT * KTOT;
        wconv_bf16_kernel<<<(n + 255) / 256, 256, 0, stream>>>(wgt, wbf);
    }
    // 2) gather + WMMA GEMM
    {
        dim3 grid(NPTS / NTILE, 1, 1);           // 1024 blocks
        spconv_wmma_kernel<<<grid, 512, 0, stream>>>(feat, wbf, hidx, widx, outp);
    }
}
